// Seq2seq_37108517438263
// MI455X (gfx1250) — matrix operand order repaired and checked by reading, not compiled, not run on hardware
//
#include <hip/hip_runtime.h>
#include <hip/hip_bf16.h>

#define EMB   128
#define HID   256
#define VOCAB 32000
#define BB    16
#define LS    256
#define LT    128
#define G4H   1024  // 4*HID

typedef __attribute__((ext_vector_type(16))) __bf16       v16bf;
typedef __attribute__((ext_vector_type(8)))  float        v8f;
typedef __attribute__((ext_vector_type(4)))  unsigned int v4u;

__device__ __forceinline__ float sigm(float x) { return 1.0f / (1.0f + __expf(-x)); }

// ============ WMMA fragment helpers (CDNA5 ISA 7.12.2, wave32) ============
// A 16x32 bf16 from row-major source: lane m=lane&15; one contiguous 16-elem run
// at K = hf.. (hf = 16*(lane>=16)) -> 2x b128 loads.
__device__ __forceinline__ v16bf frag_a(const __bf16* __restrict__ A, int lda) {
  const int lane = threadIdx.x & 31;
  const int m    = lane & 15;
  const int hf   = (lane & 16);  // 0 or 16
  union { v16bf v; v4u u[2]; } f;
  f.u[0] = *(const v4u*)(A + m * lda + hf);
  f.u[1] = *(const v4u*)(A + m * lda + hf + 8);
  return f.v;
}

// B 32x16 bf16 from pre-swizzled tile (tile = 512 bf16, lane-major: lane*16+e)
// -> one contiguous 32B run per lane -> 2x b128 loads.
__device__ __forceinline__ v16bf frag_b_swz(const __bf16* __restrict__ tile) {
  const int lane = threadIdx.x & 31;
  union { v16bf v; v4u u[2]; } f;
  const v4u* p = (const v4u*)tile + lane * 2;
  f.u[0] = p[0];
  f.u[1] = p[1];
  return f.v;
}

// B 32x16 bf16 where source is transposed row-major (Bt[n][k]): lane needs
// k = hf + e (e=0..15) of row n -> one contiguous 32B run -> 2x b128 loads.
__device__ __forceinline__ v16bf frag_bT(const __bf16* __restrict__ Bt, int ldt) {
  const int lane = threadIdx.x & 31;
  const int n    = lane & 15;
  const int hf   = (lane & 16);  // 0 or 16
  union { v16bf v; v4u u[2]; } f;
  const v4u* p = (const v4u*)(Bt + n * ldt + hf);
  f.u[0] = p[0];
  f.u[1] = p[1];
  return f.v;
}

__device__ __forceinline__ v8f wmma_bf16(v16bf a, v16bf b, v8f c) {
  return __builtin_amdgcn_wmma_f32_16x16x32_bf16(false, a, false, b, (short)0, c, false, false);
}

// ============ f32 -> bf16 convert into fragment-swizzled B layout ============
__global__ void k_cvt_swzB(const float* __restrict__ src, __bf16* __restrict__ dst,
                           int K, int N) {
  const size_t total = (size_t)K * N;
  for (size_t i = (size_t)blockIdx.x * blockDim.x + threadIdx.x; i < total;
       i += (size_t)gridDim.x * blockDim.x) {
    const int k  = (int)(i / N);
    const int n  = (int)(i % N);
    const int kl = k & 31;
    const size_t o = ((size_t)(n >> 4) * (K >> 5) + (k >> 5)) * 512 +
                     ((kl & 16) + (n & 15)) * 16 + (kl & 15);
    dst[o] = (__bf16)src[i];
  }
}

// ============ embedding gather + input projection ============
__global__ __launch_bounds__(128) void k_embed_proj(
    const int* __restrict__ tok, const float* __restrict__ emb,
    const float* __restrict__ Wk, const float* __restrict__ bias,
    float* __restrict__ Z) {
  __shared__ __attribute__((aligned(16))) __bf16 As[16 * EMB];   // 4 KB row-major
  __shared__ __attribute__((aligned(16))) __bf16 Bs[16 * 512];   // 16 KB swizzled
  const int m0  = blockIdx.x * 16;
  const int n0  = blockIdx.y * 64;
  const int tid = threadIdx.x;
  for (int i = tid; i < 16 * EMB; i += 128) {
    int r = i >> 7, k = i & 127;
    As[r * EMB + k] = (__bf16)emb[(size_t)tok[m0 + r] * EMB + k];
  }
  for (int i = tid; i < EMB * 64; i += 128) {
    int k = i >> 6, c = i & 63;  // coalesced read of Wk
    int o = ((c >> 4) * 4 + (k >> 5)) * 512 + ((k & 16) + (c & 15)) * 16 + (k & 15);
    Bs[o] = (__bf16)Wk[k * G4H + n0 + c];
  }
  __syncthreads();
  const int w = tid >> 5;
  v8f acc = {};
#pragma unroll
  for (int kc = 0; kc < EMB / 32; ++kc) {
    v16bf a = frag_a(As + kc * 32, EMB);
    v16bf b = frag_b_swz(&Bs[(w * 4 + kc) * 512]);
    acc = wmma_bf16(a, b, acc);
  }
  const int lane = tid & 31;
  const int n    = lane & 15;
  const int mh   = (lane & 16) ? 8 : 0;
  const int col  = n0 + w * 16 + n;
#pragma unroll
  for (int r = 0; r < 8; ++r)
    Z[(size_t)(m0 + r + mh) * G4H + col] = acc[r] + bias[col];
}

// ============ LSTM recurrence: one persistent 512-thread workgroup ============
__global__ __launch_bounds__(512) void k_lstm(
    const float* __restrict__ Z,        // [B][T][4H] f32 (bias folded)
    const __bf16* __restrict__ WrSwz,   // swizzled [64 n-tiles][8 k-tiles][512]
    const float* __restrict__ h0, const float* __restrict__ c0,  // [16*256] or null
    int T,
    __bf16* __restrict__ outPrev,       // h_{t-1} -> [b][T][HID], may be null
    __bf16* __restrict__ outA, int ldA, // h_t -> row (b*T+t)*ldA
    __bf16* __restrict__ outAT,         // h_t transposed -> [b][HID][T], may be null
    float* __restrict__ hT, float* __restrict__ cT) {
  extern __shared__ __attribute__((aligned(16))) char smem[];
  __bf16* hs = (__bf16*)smem;              // [16][256]   8 KB
  float*  hf = (float*)(smem + 8 * 1024);  // [16][256]  16 KB
  float*  cs = (float*)(smem + 24 * 1024); // [16][256]  16 KB
  float*  zb = (float*)(smem + 40 * 1024); // [16][1024] 64 KB
  const int tid = threadIdx.x;
  for (int i = tid; i < BB * HID; i += 512) {
    float hv = h0 ? h0[i] : 0.0f;
    float cv = c0 ? c0[i] : 0.0f;
    hs[i] = (__bf16)hv; hf[i] = hv; cs[i] = cv;
  }
  __syncthreads();
  const int w    = tid >> 5;  // wave owns columns [w*64, w*64+64)
  const int lane = tid & 31;
  const int n    = lane & 15;
  const int mh   = (lane & 16) ? 8 : 0;
  for (int t = 0; t < T; ++t) {
    if (outPrev) {
      for (int i = tid; i < BB * HID; i += 512) {
        int b = i >> 8, j = i & 255;
        outPrev[((size_t)b * T + t) * HID + j] = hs[i];
      }
    }
    v8f acc[4] = {};
#pragma unroll
    for (int kc = 0; kc < HID / 32; ++kc) {
      v16bf a = frag_a(hs + kc * 32, HID);
#pragma unroll
      for (int nt = 0; nt < 4; ++nt) {
        const int ntg = (w * 64 + nt * 16) >> 4;
        v16bf b = frag_b_swz(WrSwz + ((size_t)ntg * 8 + kc) * 512);
        acc[nt] = wmma_bf16(a, b, acc[nt]);
      }
    }
#pragma unroll
    for (int nt = 0; nt < 4; ++nt) {
      const int col = w * 64 + nt * 16 + n;
#pragma unroll
      for (int r = 0; r < 8; ++r) {
        const int brow = r + mh;  // batch index
        zb[brow * G4H + col] = acc[nt][r] + Z[((size_t)brow * T + t) * G4H + col];
      }
    }
    __syncthreads();
    for (int i = tid; i < BB * HID; i += 512) {
      int b = i >> 8, j = i & 255;
      float zi = zb[b * G4H + j];
      float zf = zb[b * G4H + HID + j];
      float zg = zb[b * G4H + 2 * HID + j];
      float zo = zb[b * G4H + 3 * HID + j];
      float cv = sigm(zf) * cs[i] + sigm(zi) * tanhf(zg);
      float hv = sigm(zo) * tanhf(cv);
      cs[i] = cv; hf[i] = hv;
      __bf16 hb = (__bf16)hv;
      hs[i] = hb;
      outA[((size_t)b * T + t) * ldA + j] = hb;
      if (outAT) outAT[((size_t)b * HID + j) * T + t] = hb;
    }
    __syncthreads();
  }
  if (hT) for (int i = tid; i < BB * HID; i += 512) hT[i] = hf[i];
  if (cT) for (int i = tid; i < BB * HID; i += 512) cT[i] = cs[i];
}

// ============ attention per batch ============
__global__ __launch_bounds__(256) void k_attn(
    const __bf16* __restrict__ SprevB, const __bf16* __restrict__ Hbf,
    const __bf16* __restrict__ HbfT, __bf16* __restrict__ featB) {
  extern __shared__ __attribute__((aligned(16))) char smem[];
  float*  sc = (float*)smem;                  // [128][256] 128 KB
  __bf16* pb = (__bf16*)(smem + 128 * 1024);  // [128][256]  64 KB
  const int b    = blockIdx.x;
  const int tid  = threadIdx.x;
  const int w    = tid >> 5;
  const int lane = tid & 31;
  const int n    = lane & 15;
  const int mh   = (lane & 16) ? 8 : 0;
  const __bf16* Sb  = SprevB + (size_t)b * LT * HID;
  const __bf16* Hb  = Hbf + (size_t)b * LS * HID;
  const __bf16* HTb = HbfT + (size_t)b * HID * LS;
  // scores (M=LT=128, N=LS=256, K=HID=256)
  for (int tile = w; tile < (LT / 16) * (LS / 16); tile += 8) {
    const int m0 = (tile >> 4) * 16, n0 = (tile & 15) * 16;
    v8f acc = {};
#pragma unroll
    for (int kc = 0; kc < HID / 32; ++kc) {
      v16bf a  = frag_a(Sb + m0 * HID + kc * 32, HID);
      v16bf bb = frag_bT(Hb + n0 * HID + kc * 32, HID);
      acc = wmma_bf16(a, bb, acc);
    }
#pragma unroll
    for (int r = 0; r < 8; ++r) sc[(m0 + r + mh) * LS + n0 + n] = acc[r];
  }
  __syncthreads();
  if (tid < LT) {
    float mx = -1e30f;
    for (int s = 0; s < LS; ++s) mx = fmaxf(mx, sc[tid * LS + s]);
    float sum = 0.0f;
    for (int s = 0; s < LS; ++s) sum += __expf(sc[tid * LS + s] - mx);
    float inv = 1.0f / sum;
    for (int s = 0; s < LS; ++s) pb[tid * LS + s] = (__bf16)(__expf(sc[tid * LS + s] - mx) * inv);
  }
  __syncthreads();
  // A = P @ H (M=LT, N=HID=256, K=LS=256)
  for (int tile = w; tile < (LT / 16) * (HID / 16); tile += 8) {
    const int m0 = (tile >> 4) * 16, n0 = (tile & 15) * 16;
    v8f acc = {};
#pragma unroll
    for (int kc = 0; kc < LS / 32; ++kc) {
      v16bf a  = frag_a(pb + m0 * LS + kc * 32, LS);
      v16bf bb = frag_bT(HTb + n0 * LS + kc * 32, LS);
      acc = wmma_bf16(a, bb, acc);
    }
#pragma unroll
    for (int r = 0; r < 8; ++r)
      featB[((size_t)b * LT + m0 + r + mh) * (2 * HID) + HID + n0 + n] = (__bf16)acc[r];
  }
}

// ============ output GEMM: logits = feat @ Wout + bout ============
// A tile staged via CDNA5 async-to-LDS (ASYNCcnt path): 64 KB raw bf16 copy,
// no VGPR round trip. Each lane issues GV-mode global_load_async_to_lds_b128
// with its LDS byte offset (low 32 bits of the shared pointer) as VDST.
__global__ __launch_bounds__(256) void k_out_gemm(
    const __bf16* __restrict__ featB, const __bf16* __restrict__ WoutSwz,
    const float* __restrict__ bout, float* __restrict__ logits) {
  __shared__ __attribute__((aligned(16))) __bf16 As[64 * 512];  // 64 KB
  const int m0  = blockIdx.x * 64;
  const int nt0 = blockIdx.y * 8;
  const int tid = threadIdx.x;
  {
    const char* src = (const char*)(featB + (size_t)m0 * 512);
    const unsigned lds_base = (unsigned)(unsigned long long)(void*)&As[0];
#pragma unroll 4
    for (int i = tid; i < 64 * 512 / 8; i += 256) {
      const unsigned lds_addr = lds_base + (unsigned)i * 16u;
      const char* g = src + (size_t)i * 16;
      asm volatile("global_load_async_to_lds_b128 %0, %1, off"
                   :: "v"(lds_addr), "v"(g) : "memory");
    }
    asm volatile("s_wait_asynccnt 0x0" ::: "memory");
  }
  __syncthreads();
  const int w  = tid >> 5;
  const int nt = nt0 + w;
  const __bf16* Bb = WoutSwz + (size_t)nt * 16 * 512;
  v8f acc[4] = {};
#pragma unroll
  for (int kc = 0; kc < 16; ++kc) {
    if (kc + 1 < 16) __builtin_prefetch(Bb + (kc + 1) * 512, 0, 1);
    v16bf b = frag_b_swz(Bb + kc * 512);
#pragma unroll
    for (int mt = 0; mt < 4; ++mt) {
      v16bf a = frag_a(As + mt * 16 * 512 + kc * 32, 512);
      acc[mt] = wmma_bf16(a, b, acc[mt]);
    }
  }
  const int lane = tid & 31;
  const int n    = lane & 15;
  const int mh   = (lane & 16) ? 8 : 0;
  const int col  = nt * 16 + n;
  const float bv = bout[col];
#pragma unroll
  for (int mt = 0; mt < 4; ++mt)
#pragma unroll
    for (int r = 0; r < 8; ++r)
      logits[(size_t)(m0 + mt * 16 + r + mh) * VOCAB + col] = acc[mt][r] + bv;
}

// ============ per-row max / sumexp over VOCAB ============
__global__ __launch_bounds__(256) void k_row_stats(
    const float* __restrict__ logits, float* __restrict__ stats) {
  __shared__ float red[256];
  const int row = blockIdx.x;
  const float* p = logits + (size_t)row * VOCAB;
  float mx = -1e30f;
  for (int i = threadIdx.x; i < VOCAB; i += 256) mx = fmaxf(mx, p[i]);
  red[threadIdx.x] = mx;
  __syncthreads();
  for (int s = 128; s > 0; s >>= 1) {
    if (threadIdx.x < s) red[threadIdx.x] = fmaxf(red[threadIdx.x], red[threadIdx.x + s]);
    __syncthreads();
  }
  mx = red[0];
  __syncthreads();
  float sum = 0.0f;
  for (int i = threadIdx.x; i < VOCAB; i += 256) sum += __expf(p[i] - mx);
  red[threadIdx.x] = sum;
  __syncthreads();
  for (int s = 128; s > 0; s >>= 1) {
    if (threadIdx.x < s) red[threadIdx.x] += red[threadIdx.x + s];
    __syncthreads();
  }
  if (threadIdx.x == 0) { stats[2 * row] = mx; stats[2 * row + 1] = red[0]; }
}

// ============ in-place softmax normalize ============
__global__ void k_softmax_norm(float* __restrict__ out, const float* __restrict__ stats) {
  const size_t total = (size_t)BB * LT * VOCAB;
  for (size_t i = (size_t)blockIdx.x * blockDim.x + threadIdx.x; i < total;
       i += (size_t)gridDim.x * blockDim.x) {
    size_t row = i / VOCAB;
    out[i] = __expf(out[i] - stats[2 * row]) / stats[2 * row + 1];
  }
}

extern "C" void kernel_launch(void* const* d_in, const int* in_sizes, int n_in,
                              void* d_out, int out_size, void* d_ws, size_t ws_size,
                              hipStream_t stream) {
  (void)in_sizes; (void)n_in; (void)out_size; (void)ws_size;
  const int*   x       = (const int*)d_in[0];
  const int*   y       = (const int*)d_in[1];
  const float* enc_emb = (const float*)d_in[2];
  const float* enc_Wk  = (const float*)d_in[3];
  const float* enc_Wr  = (const float*)d_in[4];
  const float* enc_b   = (const float*)d_in[5];
  const float* dec_emb = (const float*)d_in[6];
  const float* dec_Wk  = (const float*)d_in[7];
  const float* dec_Wr  = (const float*)d_in[8];
  const float* dec_b   = (const float*)d_in[9];
  const float* Wout    = (const float*)d_in[10];
  const float* bout    = (const float*)d_in[11];
  float* out = (float*)d_out;

  char* ws = (char*)d_ws;
  size_t off = 0;
  auto alloc = [&](size_t bytes) -> char* {
    char* p = ws + off;
    off += (bytes + 255) & ~(size_t)255;
    return p;
  };
  float*  Zenc     = (float*)alloc((size_t)BB * LS * G4H * 4);      // 16 MB
  float*  Zdec     = (float*)alloc((size_t)BB * LT * G4H * 4);      //  8 MB
  __bf16* Hbf      = (__bf16*)alloc((size_t)BB * LS * HID * 2);     //  2 MB
  __bf16* HbfT     = (__bf16*)alloc((size_t)BB * HID * LS * 2);     //  2 MB
  __bf16* SprevB   = (__bf16*)alloc((size_t)BB * LT * HID * 2);     //  1 MB
  __bf16* featB    = (__bf16*)alloc((size_t)BB * LT * 2 * HID * 2); //  2 MB
  float*  hT       = (float*)alloc((size_t)BB * HID * 4);
  float*  cT       = (float*)alloc((size_t)BB * HID * 4);
  __bf16* WrEncSwz = (__bf16*)alloc((size_t)HID * G4H * 2);         // 512 KB
  __bf16* WrDecSwz = (__bf16*)alloc((size_t)HID * G4H * 2);         // 512 KB
  __bf16* WoutSwz  = (__bf16*)alloc((size_t)2 * HID * VOCAB * 2);   //  32 MB
  float*  stats    = (float*)alloc((size_t)BB * LT * 2 * 4);

  // weight conversions into fragment-swizzled bf16
  k_cvt_swzB<<<256, 256, 0, stream>>>(enc_Wr, WrEncSwz, HID, G4H);
  k_cvt_swzB<<<256, 256, 0, stream>>>(dec_Wr, WrDecSwz, HID, G4H);
  k_cvt_swzB<<<4096, 256, 0, stream>>>(Wout, WoutSwz, 2 * HID, VOCAB);

  // embedding + input projections (bias folded)
  k_embed_proj<<<dim3(BB * LS / 16, G4H / 64), 128, 0, stream>>>(x, enc_emb, enc_Wk, enc_b, Zenc);
  k_embed_proj<<<dim3(BB * LT / 16, G4H / 64), 128, 0, stream>>>(y, dec_emb, dec_Wk, dec_b, Zdec);

  // recurrences
  const size_t lstmShm = (8 + 16 + 16 + 64) * 1024;
  k_lstm<<<1, 512, lstmShm, stream>>>(Zenc, WrEncSwz, nullptr, nullptr, LS,
                                      nullptr, Hbf, HID, HbfT, hT, cT);
  k_lstm<<<1, 512, lstmShm, stream>>>(Zdec, WrDecSwz, hT, cT, LT,
                                      SprevB, featB, 2 * HID, nullptr, nullptr, nullptr);

  // attention -> featB[:, HID:2HID]
  const size_t attnShm = (128 + 64) * 1024;
  k_attn<<<BB, 256, attnShm, stream>>>(SprevB, Hbf, HbfT, featB);

  // big GEMM + softmax over vocab
  k_out_gemm<<<dim3(BB * LT / 64, VOCAB / 128), 256, 0, stream>>>(featB, WoutSwz, bout, out);
  k_row_stats<<<BB * LT, 256, 0, stream>>>(out, stats);
  k_softmax_norm<<<8192, 256, 0, stream>>>(out, stats);
}
